// Seq2Seq_29901562315397
// MI455X (gfx1250) — matrix operand order repaired and checked by reading, not compiled, not run on hardware
//
#include <hip/hip_runtime.h>
#include <cstdint>
#include <cstddef>

#define B_DIM 2048
#define S_DIM 512
#define F_DIM 32
#define H_DIM 128
#define T_STEPS 24

typedef __attribute__((ext_vector_type(16))) __bf16 v16bf;
typedef __attribute__((ext_vector_type(8)))  float  v8f;

union Frag {
    v16bf v;
    unsigned u[8];
    unsigned short h[16];
};

__device__ __forceinline__ unsigned short f2bf(float f) {
    unsigned u = __builtin_bit_cast(unsigned, f);
    u += 0x7fffu + ((u >> 16) & 1u);
    return (unsigned short)(u >> 16);
}
__device__ __forceinline__ float bf2f(unsigned short h) {
    unsigned u = ((unsigned)h) << 16;
    return __builtin_bit_cast(float, u);
}
// branch-free fast transcendentals: v_rcp_f32 + v_exp_f32 only
__device__ __forceinline__ float fast_rcp(float x) { return __builtin_amdgcn_rcpf(x); }
__device__ __forceinline__ float fast_sigmoid(float x) {
    return fast_rcp(1.0f + __expf(-x));
}
__device__ __forceinline__ float fast_tanh(float x) {
    return __builtin_fmaf(2.0f, fast_rcp(1.0f + __expf(-2.0f * x)), -1.0f);
}
__device__ __forceinline__ v8f wmma_bf16(v16bf a, v16bf b, v8f c) {
    return __builtin_amdgcn_wmma_f32_16x16x32_bf16(false, a, false, b, (short)0, c,
                                                   false, false);
}

// A-fragment (16x32 bf16) from row-major bf16 storage (CDNA5 ISA layout).
__device__ __forceinline__ void load_afrag(Frag& f, const unsigned short* row_base,
                                           int kbase, int hi) {
    const int k0 = kbase + hi * 8;
#pragma unroll
    for (int p = 0; p < 4; ++p) {
        f.u[p]     = *(const unsigned*)(row_base + k0 + 2 * p);
        f.u[p + 4] = *(const unsigned*)(row_base + k0 + 16 + 2 * p);
    }
}

// B-fragment (32x16 bf16) for B = W^T (W row-major [n][K]).
__device__ __forceinline__ void load_bfrag(Frag& f, const unsigned short* W, int n,
                                           int rowlen, int kbase, int hi) {
    const unsigned short* p = W + (size_t)n * rowlen + kbase + hi * 8;
#pragma unroll
    for (int q = 0; q < 4; ++q) {
        f.u[q]     = *(const unsigned*)(p + 2 * q);
        f.u[q + 4] = *(const unsigned*)(p + 16 + 2 * q);
    }
}

// ---------------------------------------------------------------------------
__global__ void k_cvt_bf16(const float* __restrict__ src,
                           unsigned short* __restrict__ dst, int n) {
    int i = blockIdx.x * blockDim.x + threadIdx.x;
    if (i < n) dst[i] = f2bf(src[i]);
}

// ---------------------------------------------------------------------------
// Encoder: whole 512-step GRU scan in one kernel. 16 batch rows / workgroup,
// 8 waves; wave w owns hidden columns [16w,16w+16). h state lives in VGPRs
// (C-layout is step-invariant); bf16 copy in LDS feeds next step's A-frags.
// ---------------------------------------------------------------------------
__global__ __launch_bounds__(256) void k_encoder(
    const float* __restrict__ xb,             // [B][S][F] fp32
    const float* __restrict__ bih,            // [3H]
    const float* __restrict__ bhh,            // [3H]
    const unsigned short* __restrict__ wih,   // [3H][F] bf16
    const unsigned short* __restrict__ whh,   // [3H][H] bf16
    unsigned short* __restrict__ enc_out,     // [B][S][H] bf16
    float* __restrict__ enc_hidden)           // [B][H] fp32
{
    __shared__ unsigned short lds_hb[16][136];  // bf16 h (272B rows, 16B aligned)
    __shared__ unsigned short lds_x[16][F_DIM]; // staged x tile (bf16)

    const int tid  = threadIdx.x;
    const int wave = tid >> 5;
    const int lane = tid & 31;
    const int m    = lane & 15;
    const int hi   = lane >> 4;
    const int b0   = blockIdx.x * 16;
    const int col  = wave * 16 + m;

    for (int e = tid; e < 16 * H_DIM; e += 256) lds_hb[e >> 7][e & 127] = 0;

    // weight B-fragments resident in VGPRs for all 512 steps
    Frag bw_ih[3];
    Frag bw_hh[3][4];
#pragma unroll
    for (int g = 0; g < 3; ++g) {
        const int n = g * H_DIM + col;
        load_bfrag(bw_ih[g], wih, n, F_DIM, 0, hi);
#pragma unroll
        for (int kc = 0; kc < 4; ++kc)
            load_bfrag(bw_hh[g][kc], whh, n, H_DIM, kc * 32, hi);
    }

    const float b_ir = bih[col], b_iz = bih[H_DIM + col], b_in = bih[2 * H_DIM + col];
    const float b_hr = bhh[col], b_hz = bhh[H_DIM + col], b_hn = bhh[2 * H_DIM + col];

    float hprev[8] = {0, 0, 0, 0, 0, 0, 0, 0};  // fp32 state, lane-resident
    __syncthreads();

    for (int t = 0; t < S_DIM; ++t) {
        // stage x_t tile (16x32) as bf16
        for (int e = tid; e < 16 * F_DIM; e += 256) {
            int rm = e >> 5, k = e & 31;
            lds_x[rm][k] = f2bf(xb[((size_t)(b0 + rm) * S_DIM + t) * F_DIM + k]);
        }
        __syncthreads();

        Frag ax;
        load_afrag(ax, &lds_x[m][0], 0, hi);
        Frag ah[4];
#pragma unroll
        for (int kc = 0; kc < 4; ++kc) load_afrag(ah[kc], &lds_hb[m][0], kc * 32, hi);

        v8f ci[3], ch[3];
#pragma unroll
        for (int g = 0; g < 3; ++g) {
            v8f zi = {0, 0, 0, 0, 0, 0, 0, 0};
            ci[g] = wmma_bf16(ax.v, bw_ih[g].v, zi);
            v8f a = {0, 0, 0, 0, 0, 0, 0, 0};
#pragma unroll
            for (int kc = 0; kc < 4; ++kc) a = wmma_bf16(ah[kc].v, bw_hh[g][kc].v, a);
            ch[g] = a;
        }

#pragma unroll
        for (int r = 0; r < 8; ++r) {
            float ir = ci[0][r] + b_ir, iz = ci[1][r] + b_iz, in_ = ci[2][r] + b_in;
            float hr = ch[0][r] + b_hr, hz = ch[1][r] + b_hz, hn = ch[2][r] + b_hn;
            float rg = fast_sigmoid(ir + hr);
            float zg = fast_sigmoid(iz + hz);
            float ng = fast_tanh(in_ + rg * hn);
            hprev[r] = (1.0f - zg) * ng + zg * hprev[r];
        }
        __syncthreads();  // all A-frag reads of lds_hb done
#pragma unroll
        for (int r = 0; r < 8; ++r) lds_hb[r + hi * 8][col] = f2bf(hprev[r]);
        __syncthreads();

        // coalesced b128 store of this step's hidden state to enc_out
        {
            const int row = tid >> 4, cb = (tid & 15) * 8;
            uint4 d = *(const uint4*)&lds_hb[row][cb];
            *(uint4*)(enc_out + ((size_t)(b0 + row) * S_DIM + t) * H_DIM + cb) = d;
        }
    }

#pragma unroll
    for (int r = 0; r < 8; ++r)
        enc_hidden[(size_t)(b0 + r + hi * 8) * H_DIM + col] = hprev[r];
}

// ---------------------------------------------------------------------------
// Decoder: all 24 steps in one kernel. 16 batch rows / workgroup.
// ---------------------------------------------------------------------------
__global__ __launch_bounds__(256) void k_decoder(
    const float* __restrict__ xb,              // [B][S][F]  (for y0)
    const float* __restrict__ attn_b,          // [S]
    const float* __restrict__ dec_bih,         // [3H]
    const float* __restrict__ dec_bhh,         // [3H]
    const float* __restrict__ out1_b,          // [F]
    const float* __restrict__ out2_W,          // [1][F]
    const float* __restrict__ out2_b,          // [1]
    const unsigned short* __restrict__ attnW,  // [S][H+F] bf16
    const unsigned short* __restrict__ wih,    // [3H][H] bf16
    const unsigned short* __restrict__ whh,    // [3H][H] bf16
    const unsigned short* __restrict__ w_out1, // [F][H] bf16
    const unsigned short* __restrict__ enc_out,// [B][S][H] bf16
    const float* __restrict__ enc_hidden,      // [B][H]
    float* __restrict__ out)                   // [B][T_STEPS]
{
    __shared__ unsigned short lds_ai[16][168];  // [0:128)=h bf16, [128:160)=y bf16
    __shared__ float          lds_sc[16][S_DIM];
    __shared__ unsigned short lds_ctx[16][132];
    __shared__ float          lds_red[16][16];
    __shared__ float          lds_out[16][F_DIM];

    const int tid  = threadIdx.x;
    const int wave = tid >> 5;
    const int lane = tid & 31;
    const int m    = lane & 15;
    const int hi   = lane >> 4;
    const int b0   = blockIdx.x * 16;
    const int col  = wave * 16 + m;

    // h state in registers (same C-layout element every step)
    float hprev[8];
#pragma unroll
    for (int r = 0; r < 8; ++r) {
        float v = enc_hidden[(size_t)(b0 + r + hi * 8) * H_DIM + col];
        hprev[r] = v;
        lds_ai[r + hi * 8][col] = f2bf(v);
    }
    for (int e = tid; e < 16 * F_DIM; e += 256) {
        int r = e >> 5, c = e & 31;
        lds_ai[r][H_DIM + c] =
            f2bf(xb[((size_t)(b0 + r) * S_DIM + (S_DIM - 1)) * F_DIM + c]);
    }

    const float b_ir = dec_bih[col], b_iz = dec_bih[H_DIM + col],
                b_in = dec_bih[2 * H_DIM + col];
    const float b_hr = dec_bhh[col], b_hz = dec_bhh[H_DIM + col],
                b_hn = dec_bhh[2 * H_DIM + col];
    const float o2b = out2_b[0];
    __syncthreads();

    for (int t = 0; t < T_STEPS; ++t) {
        // ---- (a) scores(16x512) = ai(16x160) @ attnW^T, 4 N-tiles / wave ----
        Frag aai[5];
#pragma unroll
        for (int kc = 0; kc < 5; ++kc) load_afrag(aai[kc], &lds_ai[m][0], kc * 32, hi);
#pragma unroll
        for (int q = 0; q < 4; ++q) {
            const int n = (wave * 4 + q) * 16 + m;
            v8f acc = {0, 0, 0, 0, 0, 0, 0, 0};
#pragma unroll
            for (int kc = 0; kc < 5; ++kc) {
                Frag bw;
                load_bfrag(bw, attnW, n, H_DIM + F_DIM, kc * 32, hi);
                acc = wmma_bf16(aai[kc].v, bw.v, acc);
            }
            const float bb = attn_b[n];
#pragma unroll
            for (int r = 0; r < 8; ++r) lds_sc[r + hi * 8][n] = acc[r] + bb;
        }
        __syncthreads();

        // ---- (b) row softmax over S=512 (16 threads / row) ----
        {
            const int row = tid >> 4, seg = tid & 15;
            float mx = -3.4e38f;
            for (int c = 0; c < 32; ++c) mx = fmaxf(mx, lds_sc[row][seg * 32 + c]);
            lds_red[row][seg] = mx;
            __syncthreads();
            float rmax = lds_red[row][0];
#pragma unroll
            for (int j = 1; j < 16; ++j) rmax = fmaxf(rmax, lds_red[row][j]);
            __syncthreads();
            float sm = 0.0f;
            for (int c = 0; c < 32; ++c) {
                float e = __expf(lds_sc[row][seg * 32 + c] - rmax);
                lds_sc[row][seg * 32 + c] = e;
                sm += e;
            }
            lds_red[row][seg] = sm;
            __syncthreads();
            float tot = 0.0f;
#pragma unroll
            for (int j = 0; j < 16; ++j) tot += lds_red[row][j];
            const float inv = fast_rcp(tot);
            for (int c = 0; c < 32; ++c) lds_sc[row][seg * 32 + c] *= inv;
        }
        __syncthreads();

        // ---- (c) ctx[row][h] = sum_s w[s] * enc_out[b][s][h] (bf16 b128) ----
        {
            const int row = tid >> 4;
            const int hb  = (tid & 15) * 8;
            const unsigned short* eo =
                enc_out + (size_t)(b0 + row) * S_DIM * H_DIM + hb;
            float acc[8] = {0, 0, 0, 0, 0, 0, 0, 0};
            for (int s = 0; s < S_DIM; ++s) {
                const float w = lds_sc[row][s];
                uint4 d = *(const uint4*)(eo + (size_t)s * H_DIM);
                acc[0] += w * bf2f((unsigned short)(d.x & 0xffffu));
                acc[1] += w * bf2f((unsigned short)(d.x >> 16));
                acc[2] += w * bf2f((unsigned short)(d.y & 0xffffu));
                acc[3] += w * bf2f((unsigned short)(d.y >> 16));
                acc[4] += w * bf2f((unsigned short)(d.z & 0xffffu));
                acc[5] += w * bf2f((unsigned short)(d.z >> 16));
                acc[6] += w * bf2f((unsigned short)(d.w & 0xffffu));
                acc[7] += w * bf2f((unsigned short)(d.w >> 16));
            }
#pragma unroll
            for (int j = 0; j < 8; ++j) lds_ctx[row][hb + j] = f2bf(acc[j]);
        }
        __syncthreads();

        // ---- (d) GRU cell: xp = ctx@Wih^T, gh = h@Whh^T ----
        {
            Frag actx[4], ah[4];
#pragma unroll
            for (int kc = 0; kc < 4; ++kc) {
                load_afrag(actx[kc], &lds_ctx[m][0], kc * 32, hi);
                load_afrag(ah[kc], &lds_ai[m][0], kc * 32, hi);
            }
            v8f ci[3], ch[3];
#pragma unroll
            for (int g = 0; g < 3; ++g) {
                const int n = g * H_DIM + col;
                v8f a = {0, 0, 0, 0, 0, 0, 0, 0};
                v8f b = {0, 0, 0, 0, 0, 0, 0, 0};
#pragma unroll
                for (int kc = 0; kc < 4; ++kc) {
                    Frag bwi, bwh;
                    load_bfrag(bwi, wih, n, H_DIM, kc * 32, hi);
                    load_bfrag(bwh, whh, n, H_DIM, kc * 32, hi);
                    a = wmma_bf16(actx[kc].v, bwi.v, a);
                    b = wmma_bf16(ah[kc].v, bwh.v, b);
                }
                ci[g] = a;
                ch[g] = b;
            }
#pragma unroll
            for (int r = 0; r < 8; ++r) {
                float ir = ci[0][r] + b_ir, iz = ci[1][r] + b_iz, in_ = ci[2][r] + b_in;
                float hr = ch[0][r] + b_hr, hz = ch[1][r] + b_hz, hn = ch[2][r] + b_hn;
                float rg = fast_sigmoid(ir + hr);
                float zg = fast_sigmoid(iz + hz);
                float ng = fast_tanh(in_ + rg * hn);
                hprev[r] = (1.0f - zg) * ng + zg * hprev[r];
            }
            __syncthreads();  // A-frag reads of lds_ai done
#pragma unroll
            for (int r = 0; r < 8; ++r) lds_ai[r + hi * 8][col] = f2bf(hprev[r]);
        }
        __syncthreads();

        // ---- (e) out(16x32) = h_new @ out1_W^T + out1_b (waves 0,1) ----
        if (wave < 2) {
            Frag ahn[4];
#pragma unroll
            for (int kc = 0; kc < 4; ++kc) load_afrag(ahn[kc], &lds_ai[m][0], kc * 32, hi);
            const int n = wave * 16 + m;  // 0..31
            v8f acc = {0, 0, 0, 0, 0, 0, 0, 0};
#pragma unroll
            for (int kc = 0; kc < 4; ++kc) {
                Frag bw;
                load_bfrag(bw, w_out1, n, H_DIM, kc * 32, hi);
                acc = wmma_bf16(ahn[kc].v, bw.v, acc);
            }
            const float bb = out1_b[n];
#pragma unroll
            for (int r = 0; r < 8; ++r) {
                const int row = r + hi * 8;
                const float v = acc[r] + bb;
                lds_ai[row][H_DIM + n] = f2bf(v);  // y carry (pre-relu)
                lds_out[row][n] = fmaxf(v, 0.0f);  // relu'd for head
            }
        }
        __syncthreads();

        // ---- (f) final head: relu(relu(out) @ out2_W^T + out2_b) ----
        if (tid < 16) {
            float s = 0.0f;
#pragma unroll
            for (int f = 0; f < F_DIM; ++f) s += lds_out[tid][f] * out2_W[f];
            out[(size_t)(b0 + tid) * T_STEPS + t] = fmaxf(s + o2b, 0.0f);
        }
        __syncthreads();
    }
}

// ---------------------------------------------------------------------------
extern "C" void kernel_launch(void* const* d_in, const int* in_sizes, int n_in,
                              void* d_out, int out_size, void* d_ws, size_t ws_size,
                              hipStream_t stream) {
    (void)in_sizes; (void)n_in; (void)out_size; (void)ws_size;

    const float* xb       = (const float*)d_in[0];
    const float* enc_Wih  = (const float*)d_in[1];
    const float* enc_Whh  = (const float*)d_in[2];
    const float* enc_bih  = (const float*)d_in[3];
    const float* enc_bhh  = (const float*)d_in[4];
    const float* attn_W   = (const float*)d_in[5];
    const float* attn_b   = (const float*)d_in[6];
    const float* dec_Wih  = (const float*)d_in[7];
    const float* dec_Whh  = (const float*)d_in[8];
    const float* dec_bih  = (const float*)d_in[9];
    const float* dec_bhh  = (const float*)d_in[10];
    const float* out1_W   = (const float*)d_in[11];
    const float* out1_b   = (const float*)d_in[12];
    const float* out2_W   = (const float*)d_in[13];
    const float* out2_b   = (const float*)d_in[14];
    float* out            = (float*)d_out;

    char* ws = (char*)d_ws;
    size_t off = 0;
    auto carve = [&](size_t bytes) -> char* {
        char* p = ws + off;
        off = (off + bytes + 255) & ~(size_t)255;
        return p;
    };
    unsigned short* enc_out_bf = (unsigned short*)carve((size_t)B_DIM * S_DIM * H_DIM * 2);
    float*          enc_hidden = (float*)carve((size_t)B_DIM * H_DIM * 4);
    unsigned short* wih_e = (unsigned short*)carve((size_t)3 * H_DIM * F_DIM * 2);
    unsigned short* whh_e = (unsigned short*)carve((size_t)3 * H_DIM * H_DIM * 2);
    unsigned short* wattn = (unsigned short*)carve((size_t)S_DIM * (H_DIM + F_DIM) * 2);
    unsigned short* wih_d = (unsigned short*)carve((size_t)3 * H_DIM * H_DIM * 2);
    unsigned short* whh_d = (unsigned short*)carve((size_t)3 * H_DIM * H_DIM * 2);
    unsigned short* wout1 = (unsigned short*)carve((size_t)F_DIM * H_DIM * 2);

    auto cvt = [&](const float* src, unsigned short* dst, int n) {
        k_cvt_bf16<<<(n + 255) / 256, 256, 0, stream>>>(src, dst, n);
    };
    cvt(enc_Wih, wih_e, 3 * H_DIM * F_DIM);
    cvt(enc_Whh, whh_e, 3 * H_DIM * H_DIM);
    cvt(attn_W,  wattn, S_DIM * (H_DIM + F_DIM));
    cvt(dec_Wih, wih_d, 3 * H_DIM * H_DIM);
    cvt(dec_Whh, whh_d, 3 * H_DIM * H_DIM);
    cvt(out1_W,  wout1, F_DIM * H_DIM);

    k_encoder<<<B_DIM / 16, 256, 0, stream>>>(xb, enc_bih, enc_bhh, wih_e, whh_e,
                                              enc_out_bf, enc_hidden);

    k_decoder<<<B_DIM / 16, 256, 0, stream>>>(xb, attn_b, dec_bih, dec_bhh, out1_b,
                                              out2_W, out2_b, wattn, wih_d, whh_d,
                                              wout1, enc_out_bf, enc_hidden, out);
}
